// EnergyConditionedEquivariantAtomAttention_80805514707448
// MI455X (gfx1250) — matrix operand order repaired and checked by reading, not compiled, not run on hardware
//
#include <hip/hip_runtime.h>

typedef __attribute__((ext_vector_type(16))) _Float16 v16h;
typedef __attribute__((ext_vector_type(8)))  float    v8f;
typedef __attribute__((ext_vector_type(4)))  unsigned int v4u;
typedef __attribute__((ext_vector_type(8)))  int      v8i;
typedef __attribute__((ext_vector_type(4)))  int      v4i;

#define B_     16
#define N_     64
#define E_     128
#define NS_    64
#define NV_    32
#define INV_   96
#define ZE_    32
#define NRBF_  32
#define EDIM_  16
#define DNODE_ 160
#define TPW_   9216
#define CUT_   6.0f
#define SQRT3_ 1.7320508075688772f
#define ALPHA_ 0.10206207261596575f   /* 1/sqrt(96) */
#define PI_    3.14159265358979323846f

#if defined(__has_builtin)
#if __has_builtin(__builtin_amdgcn_tensor_load_to_lds) && __has_builtin(__builtin_amdgcn_s_wait_tensorcnt)
#define HAVE_TDM 1
#endif
#endif
#ifndef HAVE_TDM
#define HAVE_TDM 0
#endif

__device__ __forceinline__ float silu_f(float x){ return x / (1.f + __expf(-x)); }
__device__ __forceinline__ float sigm_f(float x){ return 1.f / (1.f + __expf(-x)); }

__device__ __forceinline__ v8f wmma32(v16h a, v16h b, v8f c){
  return __builtin_amdgcn_wmma_f32_16x16x32_f16(false, a, false, b, (short)0, c, false, false);
}

// A-fragment (16x32 f16) from row-major f16 buffer with row stride ld.
// K order used here (same as b_frag): lane m=lane&15, hi=lane>>4; element j -> K = 16*hi + j
__device__ __forceinline__ v16h a_frag(const _Float16* __restrict__ A, int ld, int kbase, int lane){
  const int m = lane & 15, hi = lane >> 4;
  v16h a;
#pragma unroll
  for (int j = 0; j < 16; ++j){
    int k = kbase + (hi << 4) + j;
    a[j] = A[m * ld + k];
  }
  return a;
}

// B-fragment (32x16 f16) from row-major W (K rows x ld cols).
// K order used here (same as a_frag): lane n=lane&15, hi=lane>>4; element j -> K = 16*hi + j
__device__ __forceinline__ v16h b_frag(const _Float16* __restrict__ W, int ld, int kbase, int nbase, int lane){
  const int n = lane & 15, hi = lane >> 4;
  v16h b;
#pragma unroll
  for (int j = 0; j < 16; ++j){
    int k = kbase + (hi << 4) + j;
    b[j] = W[k * ld + nbase + n];
  }
  return b;
}

// -------- Tensor Data Mover: 1-row 2D descriptor, contiguous f16 region ----
// D# group0: count=1, lds_addr, global_addr[56:0], type=2 ("image")
// D# group1: data_size=1 (2B), tensor_dim0=n, tensor_dim1=1, tile_dim0=n,
//            tile_dim1=1, tensor_dim0_stride=n
__device__ __forceinline__ void tdm_load_f16(void* lds_dst, const void* gsrc, unsigned int n){
#if HAVE_TDM
  const unsigned long long ga = (unsigned long long)(uintptr_t)gsrc;
  const unsigned int lds_addr = (unsigned int)(uintptr_t)lds_dst;
  v4u g0;
  g0[0] = 1u;                                   // count=1
  g0[1] = lds_addr;                             // lds_addr (bytes)
  g0[2] = (unsigned int)ga;                     // global_addr[31:0]
  g0[3] = (unsigned int)((ga >> 32) & 0x01FFFFFFu) | (2u << 30);  // addr[56:32] | type=2
  v8i g1;
  g1[0] = (int)(1u << 16);                      // data_size=1 -> 2 bytes
  g1[1] = (int)((n & 0xFFFFu) << 16);           // tensor_dim0[15:0]
  g1[2] = (int)(((n >> 16) & 0xFFFFu) | (1u << 16)); // tensor_dim0[31:16] | tensor_dim1=1
  g1[3] = (int)((n & 0xFFFFu) << 16);           // tile_dim0 (n <= 65535)
  g1[4] = 1;                                    // tile_dim1=1, tile_dim2=0
  g1[5] = (int)n;                               // tensor_dim0_stride[31:0]
  g1[6] = 0;
  g1[7] = 0;
  v4i gz = {};
#if defined(__clang_major__) && (__clang_major__ >= 23)
  v8i gz8 = {};
  __builtin_amdgcn_tensor_load_to_lds(g0, g1, gz, gz, gz8, 0);
#else
  __builtin_amdgcn_tensor_load_to_lds(g0, g1, gz, gz, 0);
#endif
#else
  (void)lds_dst; (void)gsrc; (void)n;
#endif
}

__device__ __forceinline__ void tdm_wait(){
#if HAVE_TDM
  __builtin_amdgcn_s_wait_tensorcnt(0);
#endif
}

// ----------------------------------------------------------- f32 -> f16 cast
__global__ __launch_bounds__(256) void k_f16cast(const float* __restrict__ src,
                                                 _Float16* __restrict__ dst, int n)
{
  int i = blockIdx.x * 256 + threadIdx.x;
  if (i < n) dst[i] = (_Float16)src[i];
}

// ------------------- pre-swizzle vw_W2 into WMMA B-fragment-major f16 layout
// pack[((ct*4+ks)*32 + lane)*16 + j] = W2[(ks*32 + (lane>>4)*16 + j)*9216 + ct*16 + (lane&15)]
__global__ __launch_bounds__(128) void k_w2pack(const float* __restrict__ W2,
                                                _Float16* __restrict__ pack)
{
  const int ct = blockIdx.x;        // 0..575
  const int ks = blockIdx.y;        // 0..3
  const int t = threadIdx.x;
  for (int idx = t; idx < 512; idx += 128){
    int lane = idx >> 4, j = idx & 15;
    int hi = lane >> 4, nlow = lane & 15;
    int k = ks*32 + hi*16 + j;
    pack[((size_t)(ct*4 + ks)*32)*16 + idx] =
        (_Float16)W2[(size_t)k * TPW_ + ct*16 + nlow];
  }
}

// ---------------------------------------------------------------- features
__global__ __launch_bounds__(128) void k_features(
    const float* __restrict__ h_full, const int* __restrict__ z,
    const float* __restrict__ pos, const unsigned char* __restrict__ mask,
    const float* __restrict__ z_emb,
    float* __restrict__ invnei, float* __restrict__ invabs,
    _Float16* __restrict__ vin_h, float* __restrict__ y1g,
    float* __restrict__ xdoty, float* __restrict__ gscale)
{
  const int row = blockIdx.x;            // b*64 + n
  const int b = row >> 6, n = row & 63;
  const int t = threadIdx.x;
  __shared__ float y1_s[3], r_s;
  const float* hb = h_full + (size_t)row * DNODE_;
  if (t == 0){
    float rx = pos[row*3+0] - pos[(b*N_)*3+0];
    float ry = pos[row*3+1] - pos[(b*N_)*3+1];
    float rz = pos[row*3+2] - pos[(b*N_)*3+2];
    float r  = sqrtf(rx*rx + ry*ry + rz*rz + 1e-12f);
    float iv = 1.f / fmaxf(r, 1e-8f);
    y1_s[0] = SQRT3_*rx*iv; y1_s[1] = SQRT3_*ry*iv; y1_s[2] = SQRT3_*rz*iv;
    r_s = r;
    y1g[row*3+0] = y1_s[0]; y1g[row*3+1] = y1_s[1]; y1g[row*3+2] = y1_s[2];
    float inside = (r <= CUT_) ? 1.f : 0.f;
    float cw = 0.5f * (cosf(PI_ * r / CUT_) + 1.f) * inside;
    float valid = (mask[row] != 0 && r <= CUT_ && n > 0) ? 1.f : 0.f;
    gscale[row] = cw * valid;
  }
  __syncthreads();
  if (t < 64){
    float xs = hb[t];
    invnei[(size_t)row*INV_ + t] = xs;
    if (n == 0) invabs[b*INV_ + t] = xs;
  }
  if (t < 32){
    float v0 = hb[NS_ + t*3+0], v1 = hb[NS_ + t*3+1], v2 = hb[NS_ + t*3+2];
    float vn = sqrtf((v0*v0 + v1*v1 + v2*v2) * (1.f/3.f) + 1e-8f);
    invnei[(size_t)row*INV_ + 64 + t] = vn;
    if (n == 0) invabs[b*INV_ + 64 + t] = vn;
    const float delta = CUT_ / (NRBF_ - 1);
    const float gamma = 1.f / (delta*delta + 1e-12f);
    float d = fminf(r_s, CUT_) - delta * (float)t;
    float rr = __expf(-gamma * d * d);
    int zi = z[row];
    vin_h[(size_t)row*64 + t]      = (_Float16)z_emb[zi*ZE_ + t];
    vin_h[(size_t)row*64 + 32 + t] = (_Float16)rr;
    xdoty[(size_t)row*NV_ + t] = v0*y1_s[0] + v1*y1_s[1] + v2*y1_s[2];
  }
}

// ---------------- generic WMMA dense layer, N=128; W and A TDM-staged in LDS
template<int KDIM, int ACT, int OUT16>
__global__ __launch_bounds__(128) void k_dense128(
    const _Float16* __restrict__ A, const _Float16* __restrict__ W16,
    const float* __restrict__ bias, _Float16* __restrict__ outh,
    float* __restrict__ outf)
{
  __shared__ __align__(32) _Float16 Wh[KDIM*128];
  __shared__ __align__(32) _Float16 Ah[64*KDIM];
  __shared__ float bias_s[128];
  const int tid = threadIdx.x;
  const int rowblk = blockIdx.x * 64;
#if HAVE_TDM
  if (tid == 0){
    tdm_load_f16(Wh, W16, KDIM*128);
    tdm_load_f16(Ah, A + (size_t)rowblk * KDIM, 64*KDIM);
  }
  bias_s[tid] = bias[tid];
  tdm_wait();
#else
  for (int idx = tid; idx < KDIM*128; idx += 128) Wh[idx] = W16[idx];
  for (int idx = tid; idx < 64*KDIM; idx += 128) Ah[idx] = A[(size_t)rowblk*KDIM + idx];
  bias_s[tid] = bias[tid];
#endif
  __syncthreads();
  const int lane = tid & 31, wave = tid >> 5;
  const int hi = lane >> 4;
  const int rowbase = rowblk + wave * 16;
  const _Float16* Arow = Ah + wave * 16 * KDIM;
#pragma unroll
  for (int ct = 0; ct < 8; ++ct){
    v8f acc = {};
#pragma unroll
    for (int ks = 0; ks < KDIM/32; ++ks){
      v16h a = a_frag(Arow, KDIM, ks*32, lane);
      v16h b = b_frag(Wh, 128, ks*32, ct*16, lane);
      acc = wmma32(a, b, acc);
    }
    const int col = ct*16 + (lane & 15);
#pragma unroll
    for (int r = 0; r < 8; ++r){
      float v = acc[r] + bias_s[col];
      if (ACT) v = silu_f(v);
      size_t o = (size_t)(rowbase + r + hi*8) * 128 + col;
      if (OUT16) outh[o] = (_Float16)v; else outf[o] = v;
    }
  }
}

// ------------- big GEMM: 1024 x 9216, K=128; packed-B + A tiles TDM-staged
__global__ __launch_bounds__(128) void k_tpw_gemm(
    const _Float16* __restrict__ h1, const _Float16* __restrict__ pack,
    const float* __restrict__ b2, _Float16* __restrict__ tp)
{
  __shared__ __align__(32) _Float16 Atile[16*128];
  __shared__ __align__(32) _Float16 Wtile[8192];   // 4 ctiles x 4 ksteps x 512
  const int tid = threadIdx.x;
  const int rowbase = blockIdx.x * 16;
#if HAVE_TDM
  if (tid == 0){
    tdm_load_f16(Atile, h1 + (size_t)rowbase * 128, 2048);
    tdm_load_f16(Wtile, pack + (size_t)blockIdx.y * 8192, 8192);
  }
  tdm_wait();
#else
  for (int idx = tid; idx < 2048; idx += 128)
    Atile[idx] = h1[(size_t)rowbase*128 + idx];
  for (int idx = tid; idx < 8192; idx += 128)
    Wtile[idx] = pack[(size_t)blockIdx.y*8192 + idx];
#endif
  __syncthreads();
  const int lane = tid & 31, wave = tid >> 5;
  const int hi = lane >> 4, nlow = lane & 15;
  const int nbase = (blockIdx.y * 4 + wave) * 16;
  v8f acc = {};
#pragma unroll
  for (int ks = 0; ks < 4; ++ks){
    v16h a = a_frag(Atile, 128, ks*32, lane);
    v16h b = *(const v16h*)&Wtile[(wave*4 + ks)*512 + lane*16];
    acc = wmma32(a, b, acc);
  }
  const int col = nbase + nlow;
  const float bb = b2[col];
#pragma unroll
  for (int r = 0; r < 8; ++r)
    tp[(size_t)(rowbase + r + hi*8) * TPW_ + col] = (_Float16)(acc[r] + bb);
}

// ------------------------------- per-node tensor-product contraction -> values
__global__ __launch_bounds__(160) void k_contract(
    const float* __restrict__ h_full, const _Float16* __restrict__ tp,
    const float* __restrict__ y1g, const float* __restrict__ xdoty,
    float* __restrict__ values)
{
  const int row = blockIdx.x;
  const int t = threadIdx.x;
  __shared__ float xs_s[64], xv_s[96], xdy_s[32], y1_s[3];
  const float* hb = h_full + (size_t)row * DNODE_;
  if (t < 64) xs_s[t] = hb[t];
  if (t >= 64 && t < 160) xv_s[t-64] = hb[t];
  if (t < 32) xdy_s[t] = xdoty[(size_t)row*NV_ + t];
  if (t < 3)  y1_s[t]  = y1g[row*3 + t];
  __syncthreads();
  const _Float16* tpr = tp + (size_t)row * TPW_;
  if (t < 64){
    float s = 0.f;
    for (int i = 0; i < 64; ++i) s += xs_s[i] * (float)tpr[i*64 + t];
    float s2 = 0.f;
    for (int i = 0; i < 32; ++i) s2 += xdy_s[i] * (float)tpr[7168 + i*64 + t];
    values[(size_t)row*DNODE_ + t] = ALPHA_ * (s + s2 * (1.f/SQRT3_));
  } else {
    int j = t - 64, o = j/3, c = j - o*3;
    float s = 0.f;
    for (int i = 0; i < 64; ++i) s += xs_s[i] * (float)tpr[4096 + i*32 + o];
    s *= y1_s[c];
    float s3 = 0.f;
    for (int i = 0; i < 32; ++i) s3 += xv_s[i*3 + c] * (float)tpr[6144 + i*32 + o];
    values[(size_t)row*DNODE_ + 64 + j] = ALPHA_ * (s + s3);
  }
}

// -------------------------------------- gating bases (folded b0 + inv_abs)
__global__ __launch_bounds__(128) void k_base_bn(
    const float* __restrict__ invnei, const float* __restrict__ invabs,
    const _Float16* __restrict__ vin_h, const float* __restrict__ scW0,
    const float* __restrict__ scb0, float* __restrict__ basebn)
{
  const int row = blockIdx.x; const int b = row >> 6;
  const int col = threadIdx.x;
  float s = scb0[col];
  for (int k = 0; k < 96; ++k) s += invabs[b*INV_ + k]            * scW0[k*128 + col];
  for (int k = 0; k < 96; ++k) s += invnei[(size_t)row*INV_ + k]  * scW0[(96+k)*128 + col];
  for (int k = 0; k < 64; ++k) s += (float)vin_h[(size_t)row*64+k]* scW0[(192+k)*128 + col];
  basebn[(size_t)row*128 + col] = s;
}

__global__ __launch_bounds__(128) void k_base_e(
    const float* __restrict__ e_feat, const float* __restrict__ scW0,
    float* __restrict__ basee)
{
  const int e = blockIdx.x, col = threadIdx.x;
  float s = 0.f;
  for (int k = 0; k < 16; ++k) s += e_feat[e*EDIM_ + k] * scW0[(256+k)*128 + col];
  basee[e*128 + col] = s;
}

// --------------------------- fused gate: silu(pre) -> WMMA -> silu -> sigmoid
__global__ __launch_bounds__(128) void k_gate(
    const float* __restrict__ basebn, const float* __restrict__ basee,
    const _Float16* __restrict__ scW1h, const float* __restrict__ scb1,
    const float* __restrict__ scW2, const float* __restrict__ scb2,
    const float* __restrict__ gscale, float* __restrict__ gate)
{
  __shared__ __align__(32) _Float16 W1h[128*128];
  __shared__ __align__(32) _Float16 Htile[4][16*128];
  __shared__ float basee_s[128], b1_s[128], w2_s[128];
  const int be = blockIdx.x;            // b*128 + e
  const int b = be >> 7, e = be & 127;
  const int tid = threadIdx.x;
#if HAVE_TDM
  if (tid == 0) tdm_load_f16(W1h, scW1h, 128*128);  // disjoint LDS region; overlaps with pre/silu below
#else
  for (int idx = tid; idx < 128*128; idx += 128) W1h[idx] = scW1h[idx];
#endif
  basee_s[tid] = basee[e*128 + tid];
  b1_s[tid] = scb1[tid];
  w2_s[tid] = scW2[tid];                // sc_W2 is (128,1)
  __syncthreads();                      // basee_s is read cross-wave below
  const int lane = tid & 31, wave = tid >> 5;
  const int hi = lane >> 4, nlow = lane & 15;
  const int nbase = wave * 16;
  const float* bn = basebn + (size_t)(b * N_ + nbase) * 128;
  _Float16* Ht = Htile[wave];
  for (int idx = lane; idx < 16*128; idx += 32){
    float pre = bn[idx] + basee_s[idx & 127];
    Ht[idx] = (_Float16)silu_f(pre);
  }
  tdm_wait();
  __syncthreads();
  float p[8];
#pragma unroll
  for (int r = 0; r < 8; ++r) p[r] = 0.f;
#pragma unroll
  for (int ct = 0; ct < 8; ++ct){
    v8f acc = {};
#pragma unroll
    for (int ks = 0; ks < 4; ++ks){
      v16h a   = a_frag(Ht, 128, ks*32, lane);
      v16h bfr = b_frag(W1h, 128, ks*32, ct*16, lane);
      acc = wmma32(a, bfr, acc);
    }
    const int col = ct*16 + nlow;
#pragma unroll
    for (int r = 0; r < 8; ++r){
      float h2 = silu_f(acc[r] + b1_s[col]);
      p[r] += h2 * w2_s[col];
    }
  }
#pragma unroll
  for (int r = 0; r < 8; ++r){
    p[r] += __shfl_xor(p[r], 1);
    p[r] += __shfl_xor(p[r], 2);
    p[r] += __shfl_xor(p[r], 4);
    p[r] += __shfl_xor(p[r], 8);
  }
  if (nlow == 0){
    const float b2v = scb2[0];
#pragma unroll
    for (int r = 0; r < 8; ++r){
      int n = nbase + r + hi*8;
      float g = sigm_f(p[r] + b2v) * gscale[b*N_ + n];
      gate[(size_t)be * N_ + n] = g;
    }
  }
}

// ------------------------- gated aggregation + norm + inv feats of aggregate
__global__ __launch_bounds__(192) void k_agg(
    const float* __restrict__ gate, const float* __restrict__ values,
    _Float16* __restrict__ invagg)
{
  const int be = blockIdx.x; const int b = be >> 7;
  const int t = threadIdx.x;
  __shared__ float g_s[64], agg_s[160], norm_s;
  if (t < 64) g_s[t] = gate[(size_t)be*64 + t];
  __syncthreads();
  if (t == 0){
    float s = 0.f;
    for (int n = 0; n < 64; ++n) s += g_s[n];
    norm_s = fmaxf(s, 1e-8f);
  }
  __syncthreads();
  if (t < 160){
    float s = 0.f;
    for (int n = 0; n < 64; ++n)
      s += g_s[n] * values[(size_t)(b*64 + n)*DNODE_ + t];
    agg_s[t] = s / norm_s;
  }
  __syncthreads();
  if (t < 96){
    float ia;
    if (t < 64) ia = agg_s[t];
    else {
      int o = t - 64;
      float a0 = agg_s[64+o*3], a1 = agg_s[64+o*3+1], a2 = agg_s[64+o*3+2];
      ia = sqrtf((a0*a0 + a1*a1 + a2*a2) * (1.f/3.f) + 1e-8f);
    }
    invagg[(size_t)be*96 + t] = (_Float16)ia;
  }
}

// ---------------------------------------------------------------------------
extern "C" void kernel_launch(void* const* d_in, const int* in_sizes, int n_in,
                              void* d_out, int out_size, void* d_ws, size_t ws_size,
                              hipStream_t stream)
{
  (void)in_sizes; (void)n_in; (void)out_size; (void)ws_size;
  const float*         h_full = (const float*)d_in[0];
  const int*           z      = (const int*)d_in[1];
  const float*         pos    = (const float*)d_in[2];
  const unsigned char* mask   = (const unsigned char*)d_in[3];
  const float*         e_feat = (const float*)d_in[4];
  const float*         z_emb  = (const float*)d_in[5];
  const float* vw_W0 = (const float*)d_in[6];
  const float* vw_b0 = (const float*)d_in[7];
  const float* vw_W1 = (const float*)d_in[8];
  const float* vw_b1 = (const float*)d_in[9];
  const float* vw_W2 = (const float*)d_in[10];
  const float* vw_b2 = (const float*)d_in[11];
  const float* sc_W0 = (const float*)d_in[12];
  const float* sc_b0 = (const float*)d_in[13];
  const float* sc_W1 = (const float*)d_in[14];
  const float* sc_b1 = (const float*)d_in[15];
  const float* sc_W2 = (const float*)d_in[16];
  const float* sc_b2 = (const float*)d_in[17];
  const float* out_W0 = (const float*)d_in[18];
  const float* out_b0 = (const float*)d_in[19];
  const float* out_W1 = (const float*)d_in[20];
  const float* out_b1 = (const float*)d_in[21];
  const float* out_W2 = (const float*)d_in[22];
  const float* out_b2 = (const float*)d_in[23];
  float* out = (float*)d_out;

  char* ws = (char*)d_ws;
  size_t off = 0;
  auto alloc = [&](size_t bytes) -> char* {
    char* p = ws + off;
    off += (bytes + 255) & ~(size_t)255;
    return p;
  };
  _Float16* vin_h  = (_Float16*)alloc((size_t)1024*64*2);
  float*    invnei = (float*)   alloc((size_t)1024*96*4);
  float*    invabs = (float*)   alloc((size_t)16*96*4);
  float*    y1g    = (float*)   alloc((size_t)1024*3*4);
  float*    xdoty  = (float*)   alloc((size_t)1024*32*4);
  float*    gscale = (float*)   alloc((size_t)1024*4);
  _Float16* h0vw   = (_Float16*)alloc((size_t)1024*128*2);
  _Float16* h1vw   = (_Float16*)alloc((size_t)1024*128*2);
  _Float16* tp     = (_Float16*)alloc((size_t)1024*9216*2);
  float*    basebn = (float*)   alloc((size_t)1024*128*4);
  float*    basee  = (float*)   alloc((size_t)128*128*4);
  float*    values = (float*)   alloc((size_t)1024*160*4);
  float*    gate   = (float*)   alloc((size_t)16*128*64*4);
  _Float16* invagg = (_Float16*)alloc((size_t)2048*96*2);
  _Float16* h0out  = (_Float16*)alloc((size_t)2048*128*2);
  _Float16* h1out  = (_Float16*)alloc((size_t)2048*128*2);
  // pre-converted f16 weights + fragment-packed vw_W2
  _Float16* vwW0h  = (_Float16*)alloc((size_t)64*128*2);
  _Float16* vwW1h  = (_Float16*)alloc((size_t)128*128*2);
  _Float16* scW1h  = (_Float16*)alloc((size_t)128*128*2);
  _Float16* outW0h = (_Float16*)alloc((size_t)96*128*2);
  _Float16* outW1h = (_Float16*)alloc((size_t)128*128*2);
  _Float16* outW2h = (_Float16*)alloc((size_t)128*128*2);
  _Float16* w2pack = (_Float16*)alloc((size_t)576*4*512*2);

  // one-time weight conversions / packing
  k_f16cast<<<(64*128+255)/256, 256, 0, stream>>>(vw_W0, vwW0h, 64*128);
  k_f16cast<<<(128*128+255)/256, 256, 0, stream>>>(vw_W1, vwW1h, 128*128);
  k_f16cast<<<(128*128+255)/256, 256, 0, stream>>>(sc_W1, scW1h, 128*128);
  k_f16cast<<<(96*128+255)/256, 256, 0, stream>>>(out_W0, outW0h, 96*128);
  k_f16cast<<<(128*128+255)/256, 256, 0, stream>>>(out_W1, outW1h, 128*128);
  k_f16cast<<<(128*128+255)/256, 256, 0, stream>>>(out_W2, outW2h, 128*128);
  k_w2pack<<<dim3(576,4), 128, 0, stream>>>(vw_W2, w2pack);

  k_features<<<1024, 128, 0, stream>>>(h_full, z, pos, mask, z_emb,
                                       invnei, invabs, vin_h, y1g, xdoty, gscale);
  k_base_e<<<128, 128, 0, stream>>>(e_feat, sc_W0, basee);
  k_dense128<64,1,1><<<16, 128, 0, stream>>>(vin_h, vwW0h, vw_b0, h0vw, nullptr);
  k_dense128<128,1,1><<<16, 128, 0, stream>>>(h0vw, vwW1h, vw_b1, h1vw, nullptr);
  k_base_bn<<<1024, 128, 0, stream>>>(invnei, invabs, vin_h, sc_W0, sc_b0, basebn);
  k_tpw_gemm<<<dim3(64,144), 128, 0, stream>>>(h1vw, w2pack, vw_b2, tp);
  k_contract<<<1024, 160, 0, stream>>>(h_full, tp, y1g, xdoty, values);
  k_gate<<<2048, 128, 0, stream>>>(basebn, basee, scW1h, sc_b1, sc_W2, sc_b2, gscale, gate);
  k_agg<<<2048, 192, 0, stream>>>(gate, values, invagg);
  k_dense128<96,1,1><<<32, 128, 0, stream>>>(invagg, outW0h, out_b0, h0out, nullptr);
  k_dense128<128,1,1><<<32, 128, 0, stream>>>(h0out, outW1h, out_b1, h1out, nullptr);
  k_dense128<128,0,0><<<32, 128, 0, stream>>>(h1out, outW2h, out_b2, nullptr, out);
}
